// PraxisAttention_61512521614063
// MI455X (gfx1250) — matrix operand order repaired and checked by reading, not compiled, not run on hardware
//
#include <hip/hip_runtime.h>

typedef _Float16 half4  __attribute__((ext_vector_type(4)));
typedef _Float16 half8  __attribute__((ext_vector_type(8)));
typedef _Float16 v16h   __attribute__((ext_vector_type(16)));
typedef float    v8f    __attribute__((ext_vector_type(8)));

constexpr int BATCH = 2;
constexpr int HQ = 16;
constexpr int HP = 8;      // head pairs
constexpr int S  = 2048;
constexpr int D  = 64;
constexpr int DV = 128;
constexpr int QT = 64;     // q rows per block (16 per wave-pair, 8 waves)
constexpr int KT = 32;     // keys per iteration
// 1/sqrt(64) * log2(e): run softmax in the exp2 domain (v_exp_f32 is exp2)
constexpr float SCALE2 = 0.125f * 1.44269504088896340736f;
constexpr float LAMBDA_INIT = 0.8f;
constexpr float MASKVAL = -1.0e9f;

// LDS arena (bytes). K/V tiles double-buffered; epilogue Ob overlaps dead data.
constexpr unsigned OFF_QS     = 0;                  // [2][64][72] f16 = 18432
constexpr unsigned OFF_KSa[2] = {18432, 27648};     // [2][32][72] f16 =  9216 each
constexpr unsigned OFF_VTa[2] = {36864, 47104};     // [128][40]  f16 = 10240 each
constexpr unsigned OFF_PS     = 57344;              // [8][16][40] f16 = 10240
constexpr unsigned OFF_LAM    = 67584;              // float
constexpr unsigned OFF_OB     = 0;                  // [4][16][132] f32 = 33792 (overlap)
constexpr unsigned SMEM_BYTES = 67600;

// Async b128 copy global->LDS (ASYNCcnt-tracked, no VGPR round trip).
__device__ __forceinline__ void async_cp_b128(unsigned ldsoff, const _Float16* g) {
  asm volatile("global_load_async_to_lds_b128 %0, %1, off"
               :: "v"(ldsoff), "v"(g) : "memory");
}

// Build a v16h fragment from two 8-halfword runs in LDS (each a ds_load_b128).
__device__ __forceinline__ v16h ld_frag2(const _Float16* p0, const _Float16* p1) {
  half8 a = *(const half8*)p0;
  half8 b = *(const half8*)p1;
  v16h r;
#pragma unroll
  for (int i = 0; i < 8; ++i) { r[i] = a[i]; r[i + 8] = b[i]; }
  return r;
}

// Max-reduce across each 16-lane row via DPP row rotations (pure VALU).
__device__ __forceinline__ float rowmax16(float x) {
  int t;
  t = __builtin_amdgcn_mov_dpp(__float_as_int(x), 0x128, 0xf, 0xf, true); // row_ror:8
  x = fmaxf(x, __int_as_float(t));
  t = __builtin_amdgcn_mov_dpp(__float_as_int(x), 0x124, 0xf, 0xf, true); // row_ror:4
  x = fmaxf(x, __int_as_float(t));
  t = __builtin_amdgcn_mov_dpp(__float_as_int(x), 0x122, 0xf, 0xf, true); // row_ror:2
  x = fmaxf(x, __int_as_float(t));
  t = __builtin_amdgcn_mov_dpp(__float_as_int(x), 0x121, 0xf, 0xf, true); // row_ror:1
  x = fmaxf(x, __int_as_float(t));
  return x;
}

template <bool PRE>
__launch_bounds__(256)
__global__ void diffattn_kernel(const float* __restrict__ q,
                                const float* __restrict__ kk,
                                const float* __restrict__ vv,
                                const _Float16* __restrict__ kh,
                                const _Float16* __restrict__ vth,
                                const float* __restrict__ lq1,
                                const float* __restrict__ lk1,
                                const float* __restrict__ lq2,
                                const float* __restrict__ lk2,
                                float* __restrict__ out) {
  __shared__ __align__(16) char smem[SMEM_BYTES];
  _Float16 (*Qs)[64][72]  = (_Float16(*)[64][72])(smem + OFF_QS);   // [head][row][d]
  _Float16 (*Ps)[16][40]  = (_Float16(*)[16][40])(smem + OFF_PS);   // [wave][row][key]
  float    (*Ob)[16][132] = (float(*)[16][132])(smem + OFF_OB);     // [rg][row][dv]
  float* lamS = (float*)(smem + OFF_LAM);

  const int tid  = threadIdx.x;
  const int wv   = tid >> 5;       // wave id (0..7)
  const int lane = tid & 31;
  const int col  = lane & 15;      // N / M-in-tile index
  const int hf   = lane >> 4;      // half-wave select

  const int head = wv & 1;         // which head of the pair this wave computes
  const int rg   = wv >> 1;        // q row-group (0..3)

  const int bid = blockIdx.x;
  const int qb  = bid & 31;                 // S/QT = 32
  const int hp  = (bid >> 5) & 7;
  const int b   = bid >> 8;
  const int blockQ = qb * QT;
  const int qrow0  = blockQ + rg * 16;      // first q row of this wave
  const int qLast  = qrow0 + 15;

  if (tid == 0) {
    float s1 = 0.f, s2 = 0.f;
    for (int i = 0; i < D; ++i) { s1 += lq1[i] * lk1[i]; s2 += lq2[i] * lk2[i]; }
    *lamS = __expf(s1) - __expf(s2) + LAMBDA_INIT;
  }

  // stage Q (both heads), float4 loads, scale+log2e folded, fp32 -> f16
  for (int idx = tid; idx < 2 * QT * D / 4; idx += 256) {
    int h   = idx >> 10;
    int rem = idx & 1023;
    int r   = rem >> 4;
    int d4  = (rem & 15) << 2;
    size_t g = ((size_t)(b * HQ + 2 * hp + h) * S + (blockQ + r)) * D + d4;
    float4 f = *(const float4*)(q + g);
    half4 hv = { (_Float16)(f.x * SCALE2), (_Float16)(f.y * SCALE2),
                 (_Float16)(f.z * SCALE2), (_Float16)(f.w * SCALE2) };
    *(half4*)&Qs[h][r][d4] = hv;
  }

  // Stage one K/V strip into LDS buffer `bufs`.
  auto stageTile = [&](int kt2, int bufs) {
    const int kb = kt2 * KT;
    if constexpr (PRE) {
      // 512 K chunks + 512 V chunks of 16B, 2+2 per thread, async to LDS
#pragma unroll
      for (int i = 0; i < 2; ++i) {
        int c = tid + (i << 8);
        {
          int h = c >> 8, rem = c & 255, r = rem >> 3, d8 = (rem & 7) << 3;
          const _Float16* g = kh + ((size_t)(b * HQ + 2 * hp + h) * S + kb + r) * D + d8;
          async_cp_b128(OFF_KSa[bufs] + (unsigned)(((h * 32 + r) * 72 + d8) * 2), g);
        }
        {
          int dv = c >> 2, j = (c & 3) << 3;
          const _Float16* g = vth + ((size_t)(b * HP + hp) * DV + dv) * S + kb + j;
          async_cp_b128(OFF_VTa[bufs] + (unsigned)((dv * 40 + j) * 2), g);
        }
      }
    } else {
      _Float16 (*KsT)[32][72] = (_Float16(*)[32][72])(smem + OFF_KSa[bufs]);
      _Float16 (*VtT)[40]     = (_Float16(*)[40])(smem + OFF_VTa[bufs]);
      for (int idx = tid; idx < 2 * KT * D / 4; idx += 256) {
        int h = idx >> 9, rem = idx & 511, r = rem >> 4, d4 = (rem & 15) << 2;
        size_t g = ((size_t)(b * HQ + 2 * hp + h) * S + (kb + r)) * D + d4;
        float4 f = *(const float4*)(kk + g);
        half4 hv = { (_Float16)f.x, (_Float16)f.y, (_Float16)f.z, (_Float16)f.w };
        *(half4*)&KsT[h][r][d4] = hv;
      }
      for (int idx = tid; idx < KT * DV / 4; idx += 256) {
        int r = idx >> 5, dv4 = (idx & 31) << 2;
        size_t g = ((size_t)(b * HP + hp) * S + (kb + r)) * DV + dv4;
        float4 f = *(const float4*)(vv + g);
        VtT[dv4 + 0][r] = (_Float16)f.x;
        VtT[dv4 + 1][r] = (_Float16)f.y;
        VtT[dv4 + 2][r] = (_Float16)f.z;
        VtT[dv4 + 3][r] = (_Float16)f.w;
      }
    }
  };

  // per-wave flash state (ghostmax in exp2 domain: m starts 0, l starts 1)
  float m[8], l[8];
  v8f acc[8];
#pragma unroll
  for (int r = 0; r < 8; ++r) { m[r] = 0.f; l[r] = 1.f; }
#pragma unroll
  for (int c = 0; c < 8; ++c)
#pragma unroll
    for (int r = 0; r < 8; ++r) acc[c][r] = 0.f;

  v16h ones;
#pragma unroll
  for (int i = 0; i < 16; ++i) ones[i] = (_Float16)1.0f;

  v16h aq[2];
  bool aq_loaded = false;

  const int nKT = (blockQ + QT) / KT;       // causal upper bound (uniform per block)
  stageTile(0, 0);                          // prologue: tile 0 into buffer 0

  for (int kt = 0; kt < nKT; ++kt) {
    const int kbase = kt * KT;
    const int buf   = kt & 1;
    __syncthreads();                        // compute on buf^1 finished -> reusable
    if (kt + 1 < nKT) {
      stageTile(kt + 1, buf ^ 1);           // prefetch next strip
      if constexpr (PRE) asm volatile("s_wait_asynccnt 0x4" ::: "memory");
    } else {
      if constexpr (PRE) asm volatile("s_wait_asynccnt 0x0" ::: "memory");
    }
    __syncthreads();                        // publish tile kt

    _Float16 (*Ksb)[32][72] = (_Float16(*)[32][72])(smem + OFF_KSa[buf]);
    _Float16 (*Vtb)[40]     = (_Float16(*)[40])(smem + OFF_VTa[buf]);

    if (!aq_loaded) {                       // Qs stable after first barrier
#pragma unroll
      for (int kb2 = 0; kb2 < 2; ++kb2) {
        const _Float16* p0 = &Qs[head][rg * 16 + col][kb2 * 32 + hf * 16];
        aq[kb2] = ld_frag2(p0, p0 + 8);
      }
      aq_loaded = true;
    }

    if (kbase <= qLast) {                   // wave-uniform causal skip
      // scores: two 16x16 tiles over a 32-key strip, contraction over d=64
      v8f c0, c1;
#pragma unroll
      for (int r = 0; r < 8; ++r) { c0[r] = 0.f; c1[r] = 0.f; }
#pragma unroll
      for (int kb2 = 0; kb2 < 2; ++kb2) {
        const _Float16* pb0 = &Ksb[head][col][kb2 * 32 + hf * 16];
        v16h b0 = ld_frag2(pb0, pb0 + 8);
        c0 = __builtin_amdgcn_wmma_f32_16x16x32_f16(false, aq[kb2], false, b0,
                                                    (short)0, c0, false, false);
        const _Float16* pb1 = &Ksb[head][16 + col][kb2 * 32 + hf * 16];
        v16h b1 = ld_frag2(pb1, pb1 + 8);
        c1 = __builtin_amdgcn_wmma_f32_16x16x32_f16(false, aq[kb2], false, b1,
                                                    (short)0, c1, false, false);
      }
      // causal mask: only the diagonal strip needs it (wave-uniform test)
      if (kbase + 31 > qrow0) {
#pragma unroll
        for (int r = 0; r < 8; ++r) {
          const int rowg = qrow0 + hf * 8 + r;
          if (kbase + col > rowg)      c0[r] = MASKVAL;
          if (kbase + 16 + col > rowg) c1[r] = MASKVAL;
        }
      }
      // online ghost-softmax update (exp2 domain); row max via DPP butterfly
      float alpha[8];
#pragma unroll
      for (int r = 0; r < 8; ++r) {
        float tm = rowmax16(fmaxf(c0[r], c1[r]));
        float mn = fmaxf(m[r], tm);
        alpha[r] = __builtin_amdgcn_exp2f(m[r] - mn);
        float p0 = __builtin_amdgcn_exp2f(c0[r] - mn);
        float p1 = __builtin_amdgcn_exp2f(c1[r] - mn);
        m[r] = mn;
        Ps[wv][hf * 8 + r][col]      = (_Float16)p0;
        Ps[wv][hf * 8 + r][16 + col] = (_Float16)p1;
      }
#pragma unroll
      for (int c8 = 0; c8 < 8; ++c8)
#pragma unroll
        for (int r = 0; r < 8; ++r) acc[c8][r] *= alpha[r];

      asm volatile("s_wait_dscnt 0x0" ::: "memory");  // same-wave LDS RAW order
      const _Float16* pa = &Ps[wv][col][hf * 16];
      v16h ap = ld_frag2(pa, pa + 8);

      // row sums of P via WMMA against all-ones B (replaces lane shuffles)
      v8f rz;
#pragma unroll
      for (int r = 0; r < 8; ++r) rz[r] = 0.f;
      v8f rsum = __builtin_amdgcn_wmma_f32_16x16x32_f16(false, ap, false, ones,
                                                        (short)0, rz, false, false);
#pragma unroll
      for (int r = 0; r < 8; ++r) l[r] = l[r] * alpha[r] + rsum[r];

#pragma unroll
      for (int c8 = 0; c8 < 8; ++c8) {
        const _Float16* pv = &Vtb[c8 * 16 + col][hf * 16];
        v16h bv = ld_frag2(pv, pv + 8);
        acc[c8] = __builtin_amdgcn_wmma_f32_16x16x32_f16(false, ap, false, bv,
                                                         (short)0, acc[c8],
                                                         false, false);
      }
    }
  }

  // ---- epilogue: combine o0 - lam*o1 across wave pairs through LDS ----
  float inv[8];
#pragma unroll
  for (int r = 0; r < 8; ++r) inv[r] = __builtin_amdgcn_rcpf(l[r]);

  __syncthreads();                 // K/V/Q staging dead; Ob may alias it now
  if (head == 1) {
    const float lam = *lamS;
#pragma unroll
    for (int c8 = 0; c8 < 8; ++c8)
#pragma unroll
      for (int r = 0; r < 8; ++r)
        Ob[rg][hf * 8 + r][c8 * 16 + col] = lam * acc[c8][r] * inv[r];
  }
  __syncthreads();
  if (head == 0) {
    const size_t obase = (size_t)(b * HP + hp) * S * DV;
#pragma unroll
    for (int c8 = 0; c8 < 8; ++c8) {
#pragma unroll
      for (int r = 0; r < 8; ++r) {
        const int qg = qrow0 + hf * 8 + r;
        const int dv = c8 * 16 + col;
        out[obase + (size_t)qg * DV + dv] =
            acc[c8][r] * inv[r] - Ob[rg][hf * 8 + r][dv];
      }
    }
  }
}

// One-time fp32 -> f16 conversion of K (same layout).
__global__ void conv_k_kernel(const float* __restrict__ kk, _Float16* __restrict__ kh) {
  const size_t i4 = (size_t)blockIdx.x * 256 + threadIdx.x;  // 1048576 float4s
  float4 f = ((const float4*)kk)[i4];
  half4 hv = { (_Float16)f.x, (_Float16)f.y, (_Float16)f.z, (_Float16)f.w };
  ((half4*)kh)[i4] = hv;
}

// One-time fp32 -> f16 transpose of V: [bh][s][dv] -> [bh][dv][s] via 32x32 LDS tiles.
__global__ void conv_vt_kernel(const float* __restrict__ vv, _Float16* __restrict__ vth) {
  __shared__ _Float16 T[32][33];
  const int tb  = blockIdx.x;
  const int sv  = tb & 63;            // S/32
  const int dvb = (tb >> 6) & 3;      // DV/32
  const int bh  = tb >> 8;            // B*HP
  const int s0  = sv * 32, dv0 = dvb * 32;
  const int ty = threadIdx.x >> 5, tx = threadIdx.x & 31;
#pragma unroll
  for (int i = 0; i < 4; ++i) {
    int s = s0 + ty + i * 8;
    T[ty + i * 8][tx] = (_Float16)vv[((size_t)bh * S + s) * DV + dv0 + tx];
  }
  __syncthreads();
#pragma unroll
  for (int i = 0; i < 4; ++i) {
    int dv = dv0 + ty + i * 8;
    vth[((size_t)bh * DV + dv) * S + s0 + tx] = T[tx][ty + i * 8];
  }
}

// Each GroupNorm group (b, head, seq-half) is a contiguous 131072-float slab.
__global__ void gn_stats_kernel(const float* __restrict__ o, float* __restrict__ stats) {
  __shared__ float r1[256], r2[256];
  const int gi = blockIdx.x;                               // 32 groups
  const size_t base = (size_t)(gi >> 4) * 2097152 + (size_t)(gi & 15) * 131072;
  const float4* p = (const float4*)(o + base);
  float s = 0.f, s2 = 0.f;
  for (int i = threadIdx.x; i < 32768; i += 256) {
    float4 x = p[i];
    s  += x.x + x.y + x.z + x.w;
    s2 += x.x * x.x + x.y * x.y + x.z * x.z + x.w * x.w;
  }
  r1[threadIdx.x] = s; r2[threadIdx.x] = s2;
  __syncthreads();
  for (int st = 128; st > 0; st >>= 1) {
    if (threadIdx.x < st) {
      r1[threadIdx.x] += r1[threadIdx.x + st];
      r2[threadIdx.x] += r2[threadIdx.x + st];
    }
    __syncthreads();
  }
  if (threadIdx.x == 0) {
    float mean = r1[0] * (1.f / 131072.f);
    float var  = r2[0] * (1.f / 131072.f) - mean * mean;
    stats[2 * gi]     = mean;
    stats[2 * gi + 1] = rsqrtf(var + 1e-5f);
  }
}

__global__ void gn_apply_kernel(float* __restrict__ o, const float* __restrict__ stats,
                                const float* __restrict__ wgt, const float* __restrict__ bias) {
  const size_t i4 = (size_t)blockIdx.x * 256 + threadIdx.x;   // 1048576 float4s
  const size_t ob4 = i4 & 524287;
  const int b  = (int)(i4 >> 19);
  const int c  = (int)(ob4 >> 9);                             // same c for all 4 lanes
  const int gi = b * 16 + (int)(ob4 >> 15);
  const float mean = stats[2 * gi];
  const float rstd = stats[2 * gi + 1];
  const float wsc  = rstd * wgt[c] * (1.0f - LAMBDA_INIT);
  const float bof  = (bias[c] - mean * rstd * wgt[c]) * (1.0f - LAMBDA_INIT);
  float4* po = (float4*)o;
  float4 x = po[i4];
  x.x = x.x * wsc + bof;
  x.y = x.y * wsc + bof;
  x.z = x.z * wsc + bof;
  x.w = x.w * wsc + bof;
  po[i4] = x;
}

extern "C" void kernel_launch(void* const* d_in, const int* in_sizes, int n_in,
                              void* d_out, int out_size, void* d_ws, size_t ws_size,
                              hipStream_t stream) {
  const float* q   = (const float*)d_in[0];
  const float* k   = (const float*)d_in[1];
  const float* v   = (const float*)d_in[2];
  const float* lq1 = (const float*)d_in[3];
  const float* lk1 = (const float*)d_in[4];
  const float* lq2 = (const float*)d_in[5];
  const float* lk2 = (const float*)d_in[6];
  const float* gw  = (const float*)d_in[7];
  const float* gb  = (const float*)d_in[8];
  float* out   = (float*)d_out;
  float* stats = (float*)d_ws;   // 64 floats at the front of the workspace

  const size_t nK  = (size_t)BATCH * HQ * S * D;   // 4194304 elements
  const size_t nV  = (size_t)BATCH * HP * S * DV;  // 4194304 elements
  const size_t need = 256 + (nK + nV) * sizeof(_Float16);

  if (ws_size >= need) {
    _Float16* kh  = (_Float16*)((char*)d_ws + 256);
    _Float16* vth = kh + nK;
    conv_k_kernel<<<(int)(nK / 1024), 256, 0, stream>>>(k, kh);
    conv_vt_kernel<<<BATCH * HP * (S / 32) * (DV / 32), 256, 0, stream>>>(v, vth);
    diffattn_kernel<true><<<BATCH * HP * (S / QT), 256, 0, stream>>>(
        q, k, v, kh, vth, lq1, lk1, lq2, lk2, out);
  } else {
    diffattn_kernel<false><<<BATCH * HP * (S / QT), 256, 0, stream>>>(
        q, k, v, nullptr, nullptr, lq1, lk1, lq2, lk2, out);
  }
  gn_stats_kernel<<<BATCH * HQ, 256, 0, stream>>>(out, stats);
  gn_apply_kernel<<<(BATCH * HQ * D * S) / 1024, 256, 0, stream>>>(out, stats, gw, gb);
}
